// MultiHeadSelfAttention_20538533609474
// MI455X (gfx1250) — matrix operand order repaired and checked by reading, not compiled, not run on hardware
//
#include <hip/hip_runtime.h>

// ---------------------------------------------------------------------------
// MI455X (gfx1250) multi-head self-attention w/ RoPE + sliding window.
// f16 WMMA for both projections and for the attention core; fp32 accumulate.
// GEMMs: 128x128 block tile, K-step 64, double-buffered async global->LDS.
// ---------------------------------------------------------------------------

#define BATCH 2
#define SEQ   2048
#define EMB   2048
#define NH    16
#define HD    128
#define WIN   128
#define QKVN  6144

typedef _Float16 h16;
typedef __attribute__((ext_vector_type(16))) _Float16 v16h;
typedef __attribute__((ext_vector_type(8)))  _Float16 v8h;
typedef __attribute__((ext_vector_type(8)))  float    v8f;
typedef __attribute__((__vector_size__(16))) int      v4i;   // async-LDS builtin ptr type

#if __has_builtin(__builtin_amdgcn_global_load_async_to_lds_b128)
#define USE_ASYNC_COPY 1
#else
#define USE_ASYNC_COPY 0
#endif

__device__ __forceinline__ v16h cat8(v8h lo, v8h hi) {
  return __builtin_shufflevector(lo, hi, 0, 1, 2, 3, 4, 5, 6, 7,
                                 8, 9, 10, 11, 12, 13, 14, 15);
}
__device__ __forceinline__ int imax(int a, int b) { return a > b ? a : b; }
__device__ __forceinline__ int imin(int a, int b) { return a < b ? a : b; }

// --------------------------- elementwise converts ---------------------------

__global__ __launch_bounds__(256) void cvt_f32_to_f16_k(
    const float* __restrict__ in, h16* __restrict__ out, long n) {
  long i = (long)blockIdx.x * blockDim.x + threadIdx.x;
  long stride = (long)gridDim.x * blockDim.x;
  for (; i < n; i += stride) out[i] = (h16)in[i];
}

// in: [K][N] f32 row-major  ->  out: [N][K] f16 row-major (B transposed)
__global__ __launch_bounds__(256) void transpose_cvt_k(
    const float* __restrict__ in, h16* __restrict__ out, int K, int N) {
  long i = (long)blockIdx.x * blockDim.x + threadIdx.x;
  long total = (long)K * N;
  if (i >= total) return;
  int kk = (int)(i / N);
  int nn = (int)(i - (long)kk * N);
  out[(long)nn * K + kk] = (h16)in[i];
}

// ------------------------------- f16 WMMA GEMM ------------------------------
// C[M,N] = A[M,K] * B[K,N], A row-major f16, B given transposed Bt[N,K] f16.
// Block: 256 thr (8 waves) -> 128x128 C tile; wave -> 32x64 (2x4 wmma tiles).
// K-step 64 per round, ping-pong LDS buffers, async copies overlap WMMA.

template <int OUT_F32>
__global__ __launch_bounds__(256) void gemm_f16_k(
    const h16* __restrict__ A, const h16* __restrict__ Bt,
    void* __restrict__ Cout, const float* __restrict__ bias,
    int M, int N, int K) {
  __shared__ __align__(16) h16 As[2][128][72];  // 64 k + 8 pad (144B rows)
  __shared__ __align__(16) h16 Bs[2][128][72];  // n-major: Bs[.][n][k]

  const int tid  = threadIdx.x;
  const int lane = tid & 31;
  const int wid  = tid >> 5;
  const int half = lane >> 4;
  const int l16  = lane & 15;
  const int wm   = wid & 3;   // 4 row-waves * 32 rows
  const int wn   = wid >> 2;  // 2 col-waves * 64 cols
  const int m0   = blockIdx.y * 128;
  const int n0   = blockIdx.x * 128;

  auto stage = [&](int bsel, int kofs) {
#pragma unroll
    for (int s = 0; s < 4; ++s) {
      int c   = tid + 256 * s;     // 1024 chunks of 16B per 128x64 tile
      int row = c >> 3;
      int k8  = (c & 7) * 8;
      const h16* ag = A  + (long)(m0 + row) * K + kofs + k8;
      const h16* bg = Bt + (long)(n0 + row) * K + kofs + k8;
#if USE_ASYNC_COPY
      __builtin_amdgcn_global_load_async_to_lds_b128(
          (v4i*)ag, (v4i*)&As[bsel][row][k8], 0, 0);
      __builtin_amdgcn_global_load_async_to_lds_b128(
          (v4i*)bg, (v4i*)&Bs[bsel][row][k8], 0, 0);
#else
      *(v8h*)&As[bsel][row][k8] = *(const v8h*)ag;
      *(v8h*)&Bs[bsel][row][k8] = *(const v8h*)bg;
#endif
    }
  };

  v8f acc[2][4];
#pragma unroll
  for (int a = 0; a < 2; ++a)
#pragma unroll
    for (int b = 0; b < 4; ++b) acc[a][b] = (v8f)(0.0f);

  const int nrounds = K >> 6;

  stage(0, 0);
#if USE_ASYNC_COPY
  asm volatile("s_wait_asynccnt 0" ::: "memory");
#endif
  __syncthreads();

  int buf = 0;
  for (int rd = 0; rd < nrounds; ++rd) {
    if (rd + 1 < nrounds) stage(buf ^ 1, (rd + 1) << 6);  // prefetch next round

#pragma unroll
    for (int k32 = 0; k32 < 2; ++k32) {
      const int kb = k32 * 32;
      v16h af[2], bf[4];
#pragma unroll
      for (int mt = 0; mt < 2; ++mt) {
        const h16* ar = &As[buf][wm * 32 + mt * 16 + l16][kb];
        af[mt] = cat8(*(const v8h*)(ar + 8 * half),
                      *(const v8h*)(ar + 16 + 8 * half));
      }
#pragma unroll
      for (int nt = 0; nt < 4; ++nt) {
        const h16* br = &Bs[buf][wn * 64 + nt * 16 + l16][kb];
        bf[nt] = cat8(*(const v8h*)(br + 8 * half),
                      *(const v8h*)(br + 16 + 8 * half));
      }
#pragma unroll
      for (int mt = 0; mt < 2; ++mt)
#pragma unroll
        for (int nt = 0; nt < 4; ++nt)
          acc[mt][nt] = __builtin_amdgcn_wmma_f32_16x16x32_f16(
              false, af[mt], false, bf[nt], (short)0, acc[mt][nt], false, false);
    }

#if USE_ASYNC_COPY
    asm volatile("s_wait_asynccnt 0" ::: "memory");
#endif
    __syncthreads();
    buf ^= 1;
  }

  // ---- epilogue (compile-time output mode; bias hoisted) ----
  float bcol[4] = {0.0f, 0.0f, 0.0f, 0.0f};
  if (OUT_F32 && bias) {
#pragma unroll
    for (int nt = 0; nt < 4; ++nt)
      bcol[nt] = bias[n0 + wn * 64 + nt * 16 + l16];
  }
#pragma unroll
  for (int mt = 0; mt < 2; ++mt)
#pragma unroll
    for (int nt = 0; nt < 4; ++nt) {
      const int col = n0 + wn * 64 + nt * 16 + l16;
#pragma unroll
      for (int r = 0; r < 8; ++r) {
        const int row = m0 + wm * 32 + mt * 16 + r + 8 * half;
        float v = acc[mt][nt][r];
        if (OUT_F32) {
          ((float*)Cout)[(long)row * N + col] = v + bcol[nt];
        } else {
          ((h16*)Cout)[(long)row * N + col] = (h16)v;
        }
      }
    }
}

// ------------------------- RoPE + QKV split/transpose -----------------------
// qkv f16 [b, l, f] with f = (h*HD + d)*3 + c (c in {q,k,v}).
// -> q,k f16 [b,h,l,d] (RoPE applied), vT f16 [b,h,d,l].

__global__ __launch_bounds__(256) void rope_split_k(
    const h16* __restrict__ qkv, h16* __restrict__ q,
    h16* __restrict__ k, h16* __restrict__ vT) {
  long idx = (long)blockIdx.x * 256 + threadIdx.x;  // BATCH*NH*SEQ*(HD/2)
  int t   = (int)(idx & 63);                        // pair index
  long r0 = idx >> 6;
  int l   = (int)(r0 & (SEQ - 1));
  int bh  = (int)(r0 >> 11);                        // b*NH + h
  int h   = bh & (NH - 1);
  int b   = bh >> 4;
  int d0  = 2 * t;

  long row = (long)b * SEQ + l;
  const h16* src = qkv + row * QKVN + (long)(h * HD + d0) * 3;
  float q0 = (float)src[0], k0 = (float)src[1], v0 = (float)src[2];
  float q1 = (float)src[3], k1 = (float)src[4], v1 = (float)src[5];

  float ang = (float)l * __powf(10000.0f, -(float)d0 * (1.0f / (float)HD));
  float sn, cs;
  __sincosf(ang, &sn, &cs);

  long qo = ((long)bh * SEQ + l) * HD + d0;
  q[qo]     = (h16)(q0 * cs - q1 * sn);
  q[qo + 1] = (h16)(q1 * cs + q0 * sn);
  k[qo]     = (h16)(k0 * cs - k1 * sn);
  k[qo + 1] = (h16)(k1 * cs + k0 * sn);

  long vo = ((long)bh * HD + d0) * SEQ + l;
  vT[vo]       = (h16)v0;
  vT[vo + SEQ] = (h16)v1;
}

// ----------------------- flash attention (sliding window) ------------------
// wave handles 16 queries of one (b,h); key steps of 32; WMMA for S and P*V.

__global__ __launch_bounds__(256) void attn_k(
    const h16* __restrict__ q, const h16* __restrict__ k,
    const h16* __restrict__ vT, h16* __restrict__ o) {
  __shared__ __align__(16) h16 Plds[8][16][40];  // per-wave P patch (D->A layout)

  const int tid  = threadIdx.x;
  const int lane = tid & 31;
  const int wid  = tid >> 5;
  const int half = lane >> 4;
  const int l16  = lane & 15;

  const int gw = blockIdx.x * 8 + wid;  // global wave id, 0..4095
  const int bh = gw >> 7;               // b*NH + h
  const int qs = (gw & 127) << 4;       // query tile start
  const int b  = bh >> 4;
  const int h  = bh & (NH - 1);

  const h16* qp = q  + (long)bh * SEQ * HD;
  const h16* kp = k  + (long)bh * SEQ * HD;
  const h16* vp = vT + (long)bh * HD * SEQ;
  const float scale = 0.08838834764831845f;  // 1/sqrt(HD)

  // Q fragments for all 4 K-steps of head dim (A-operand layout)
  v16h qf[4];
#pragma unroll
  for (int d4 = 0; d4 < 4; ++d4) {
    const h16* qr = qp + (long)(qs + l16) * HD + d4 * 32;
    qf[d4] = cat8(*(const v8h*)(qr + 16 * half),
                  *(const v8h*)(qr + 16 * half + 8));
  }

  float mrow[8], lrow[8];
  v8f oacc[8];
#pragma unroll
  for (int r = 0; r < 8; ++r) { mrow[r] = -3.0e38f; lrow[r] = 0.0f; }
#pragma unroll
  for (int n8 = 0; n8 < 8; ++n8) oacc[n8] = (v8f)(0.0f);

  for (int t = 0; t < 5; ++t) {
    int kbase = qs - WIN + 32 * t;
    if (kbase + 31 < 0) continue;  // wave-uniform

    // ---- S = Q * K^T over 32 keys (two 16-key D tiles) ----
    v8f S0 = (v8f)(0.0f), S1 = (v8f)(0.0f);
#pragma unroll
    for (int d4 = 0; d4 < 4; ++d4) {
      {
        int krow = imin(imax(kbase + l16, 0), SEQ - 1);
        const h16* kr = kp + (long)krow * HD + d4 * 32 + 16 * half;
        v16h bk = cat8(*(const v8h*)kr, *(const v8h*)(kr + 8));
        S0 = __builtin_amdgcn_wmma_f32_16x16x32_f16(false, qf[d4], false, bk,
                                                    (short)0, S0, false, false);
      }
      {
        int krow = imin(imax(kbase + 16 + l16, 0), SEQ - 1);
        const h16* kr = kp + (long)krow * HD + d4 * 32 + 16 * half;
        v16h bk = cat8(*(const v8h*)kr, *(const v8h*)(kr + 8));
        S1 = __builtin_amdgcn_wmma_f32_16x16x32_f16(false, qf[d4], false, bk,
                                                    (short)0, S1, false, false);
      }
    }

    // ---- online softmax in D-layout; stats uniform within 16-lane halves ----
#pragma unroll
    for (int r = 0; r < 8; ++r) {
      int i  = qs + r + 8 * half;
      int j0 = kbase + l16;
      int j1 = j0 + 16;
      float s0 = (j0 >= 0 && j0 <= i && j0 >= i - WIN) ? S0[r] * scale : -3.0e38f;
      float s1 = (j1 >= 0 && j1 <= i && j1 >= i - WIN) ? S1[r] * scale : -3.0e38f;

      float mx = fmaxf(s0, s1);
      mx = fmaxf(mx, __shfl_xor(mx, 1));
      mx = fmaxf(mx, __shfl_xor(mx, 2));
      mx = fmaxf(mx, __shfl_xor(mx, 4));
      mx = fmaxf(mx, __shfl_xor(mx, 8));
      float nm = fmaxf(mrow[r], mx);
      float sc = __expf(mrow[r] - nm);
      float p0 = __expf(s0 - nm);
      float p1 = __expf(s1 - nm);
      float rs = p0 + p1;
      rs += __shfl_xor(rs, 1);
      rs += __shfl_xor(rs, 2);
      rs += __shfl_xor(rs, 4);
      rs += __shfl_xor(rs, 8);
      lrow[r] = lrow[r] * sc + rs;
      mrow[r] = nm;
#pragma unroll
      for (int n8 = 0; n8 < 8; ++n8) oacc[n8][r] *= sc;

      int prow = r + 8 * half;
      Plds[wid][prow][l16]      = (h16)p0;
      Plds[wid][prow][l16 + 16] = (h16)p1;
    }
    asm volatile("s_wait_dscnt 0x0" ::: "memory");  // wave-local D->A round trip

    v16h pa;
    {
      const h16* pr = &Plds[wid][l16][0];
      pa = cat8(*(const v8h*)(pr + 16 * half),
                *(const v8h*)(pr + 16 * half + 8));
    }

    // ---- O += P * V  (V^T rows are contiguous keys -> clean B frags) ----
#pragma unroll
    for (int n8 = 0; n8 < 8; ++n8) {
      const h16* vr = vp + (long)(n8 * 16 + l16) * SEQ;
      int kb = imin(imax(kbase + 16 * half, 0), SEQ - 16);
      v16h bv = cat8(*(const v8h*)(vr + kb), *(const v8h*)(vr + kb + 8));
      oacc[n8] = __builtin_amdgcn_wmma_f32_16x16x32_f16(false, pa, false, bv,
                                                        (short)0, oacc[n8],
                                                        false, false);
    }
  }

  // ---- normalize + store to o[b, l, e] ----
#pragma unroll
  for (int r = 0; r < 8; ++r) {
    float inv = 1.0f / lrow[r];
    int row = qs + r + 8 * half;
#pragma unroll
    for (int n8 = 0; n8 < 8; ++n8) {
      float v = oacc[n8][r] * inv;
      o[((long)b * SEQ + row) * EMB + h * HD + n8 * 16 + l16] = (h16)v;
    }
  }
}

// --------------------------------- launcher ---------------------------------

extern "C" void kernel_launch(void* const* d_in, const int* in_sizes, int n_in,
                              void* d_out, int out_size, void* d_ws,
                              size_t ws_size, hipStream_t stream) {
  (void)in_sizes; (void)n_in; (void)out_size; (void)ws_size;
  const float* x    = (const float*)d_in[0];
  const float* Wqkv = (const float*)d_in[1];
  const float* Wout = (const float*)d_in[2];
  const float* bout = (const float*)d_in[3];

  h16* ws    = (h16*)d_ws;                 // offsets in f16 elements
  h16* xh    = ws;                         //  8,388,608  (x  f16)
  h16* Wqkvt = ws + 8388608L;              // 12,582,912  (Wqkv^T f16)
  h16* Woutt = ws + 20971520L;             //  4,194,304  (Wout^T f16)
  h16* qkvh  = ws + 25165824L;             // 25,165,824  (qkv f16)
  h16* qh    = ws + 50331648L;             //  8,388,608
  h16* kh    = ws + 58720256L;             //  8,388,608
  h16* vTh   = ws + 67108864L;             //  8,388,608
  h16* oh    = ws + 75497472L;             //  8,388,608  (attn out f16)

  cvt_f32_to_f16_k<<<4096, 256, 0, stream>>>(x, xh, 8388608L);
  transpose_cvt_k<<<49152, 256, 0, stream>>>(Wqkv, Wqkvt, EMB, QKVN);
  transpose_cvt_k<<<16384, 256, 0, stream>>>(Wout, Woutt, EMB, EMB);

  dim3 g1(QKVN / 128, (BATCH * SEQ) / 128);
  gemm_f16_k<0><<<g1, 256, 0, stream>>>(xh, Wqkvt, (void*)qkvh, nullptr,
                                        BATCH * SEQ, QKVN, EMB);

  rope_split_k<<<16384, 256, 0, stream>>>(qkvh, qh, kh, vTh);

  attn_k<<<512, 256, 0, stream>>>(qh, kh, vTh, oh);

  dim3 g2(EMB / 128, (BATCH * SEQ) / 128);
  gemm_f16_k<1><<<g2, 256, 0, stream>>>(oh, Woutt, d_out, bout,
                                        BATCH * SEQ, EMB, EMB);
}
